// WaveVitBlock_10144712753803
// MI455X (gfx1250) — matrix operand order repaired and checked by reading, not compiled, not run on hardware
//
#include <hip/hip_runtime.h>
#include <hip/hip_bf16.h>
#include <math.h>

typedef __attribute__((ext_vector_type(16))) __bf16 bf16x16;
typedef __attribute__((ext_vector_type(8)))  float  f32x8;

#define DEV __device__ __forceinline__

// ---- WMMA helpers -------------------------------------------------------
DEV f32x8 wmma_bf16(bf16x16 a, bf16x16 b, f32x8 c) {
  return __builtin_amdgcn_wmma_f32_16x16x32_bf16(false, a, false, b, (short)0, c,
                                                 false, false);
}

// A fragment (16x32, row-major source). Caller passes pointer to
// row*ld + k0 + ((lane>>4)*8); elements 0-7 at +0, 8-15 at +16.
DEV bf16x16 ldA(const __bf16* p) {
  bf16x16 a;
  ((uint4*)&a)[0] = *(const uint4*)(p);
  ((uint4*)&a)[1] = *(const uint4*)(p + 16);
  return a;
}
DEV bf16x16 ldA_pad(const __bf16* p, bool valid) {
  bf16x16 a = {};
  if (valid) {
    ((uint4*)&a)[0] = *(const uint4*)(p);
    ((uint4*)&a)[1] = *(const uint4*)(p + 16);
  }
  return a;
}
// B fragment (32x16): lane holds K of one column row (W[n][k]).
// Caller passes pointer to n_row*ld + k0 + ((lane>>4)*8); elements 0-7 at +0, 8-15 at +16.
DEV bf16x16 ldB(const __bf16* p) {
  bf16x16 b;
  ((uint4*)&b)[0] = *(const uint4*)(p);
  ((uint4*)&b)[1] = *(const uint4*)(p + 16);
  return b;
}

// ---- packing / conversion ----------------------------------------------
__global__ void k_f2b(const float* __restrict__ X, __bf16* __restrict__ Y, long n) {
  long i = (long)blockIdx.x * blockDim.x + threadIdx.x;
  if (i < n) Y[i] = (__bf16)X[i];
}
__global__ void k_zero16(__bf16* __restrict__ Y, long n) {
  long i = (long)blockIdx.x * blockDim.x + threadIdx.x;
  if (i < n) Y[i] = (__bf16)0.f;
}
// filt_w (O=320,I=320,3,3) -> [o][(ky*3+kx)*320 + i]
__global__ void k_pack_filt(const float* __restrict__ W, __bf16* __restrict__ D) {
  long i = (long)blockIdx.x * blockDim.x + threadIdx.x;
  if (i >= 320L * 320 * 9) return;
  int kx = i % 3; long t = i / 3;
  int ky = t % 3; t /= 3;
  int ic = t % 320; int o = t / 320;
  D[(size_t)o * 2880 + (ky * 3 + kx) * 320 + ic] = (__bf16)W[i];
}
// kve_w (O=320,I=320,2,2) -> [o][(ky*2+kx)*320 + i]
__global__ void k_pack_kve(const float* __restrict__ W, __bf16* __restrict__ D) {
  long i = (long)blockIdx.x * blockDim.x + threadIdx.x;
  if (i >= 320L * 320 * 4) return;
  int kx = i % 2; long t = i / 2;
  int ky = t % 2; t /= 2;
  int ic = t % 320; int o = t / 320;
  D[(size_t)o * 1280 + (ky * 2 + kx) * 320 + ic] = (__bf16)W[i];
}
// proj_w (320,400) -> (320,416) zero-padded K
__global__ void k_pack_proj(const float* __restrict__ W, __bf16* __restrict__ D) {
  long i = (long)blockIdx.x * blockDim.x + threadIdx.x;
  if (i >= 320L * 416) return;
  int k = i % 416, n = i / 416;
  D[i] = (k < 400) ? (__bf16)W[n * 400 + k] : (__bf16)0.f;
}

// ---- LayerNorm (C=320), one wave32 per row, bf16 out --------------------
__global__ void k_ln(const float* __restrict__ X, const float* __restrict__ g,
                     const float* __restrict__ b, __bf16* __restrict__ O, int M) {
  const int C = 320;
  int lane = threadIdx.x & 31;
  int row = blockIdx.x * 8 + (threadIdx.x >> 5);
  if (row >= M) return;
  const float* xr = X + (size_t)row * C;
  float s = 0.f, s2 = 0.f;
  for (int i = lane; i < C; i += 32) { float t = xr[i]; s += t; s2 += t * t; }
  for (int o = 16; o; o >>= 1) { s += __shfl_xor(s, o); s2 += __shfl_xor(s2, o); }
  float mean = s * (1.f / C);
  float var = s2 * (1.f / C) - mean * mean;
  float inv = rsqrtf(var + 1e-5f);
  __bf16* orow = O + (size_t)row * C;
  for (int i = lane; i < C; i += 32)
    orow[i] = (__bf16)((xr[i] - mean) * inv * g[i] + b[i]);
}

// Pipelined standard K-loop: prefetch next step's fragments before WMMAs.
#define GEMM_CORE(A_, W_, M_, K_)                                              \
  const int lane = threadIdx.x & 31, wv = threadIdx.x >> 5;                    \
  const int l15 = lane & 15, ka = (lane >> 4) * 8, kb = (lane >> 4) * 8;       \
  const int mh = (lane >> 4) * 8;                                              \
  const int m0 = (blockIdx.x * 8 + wv) * 32;                                   \
  const int n0 = blockIdx.y * 64;                                              \
  if (m0 >= (M_)) return;                                                      \
  const __bf16* pa0 = (A_) + (size_t)(m0 + l15) * (K_) + ka;                   \
  const __bf16* pa1 = pa0 + (size_t)16 * (K_);                                 \
  const __bf16* pb0 = (W_) + (size_t)(n0 + l15) * (K_) + kb;                   \
  const __bf16* pb1 = pb0 + (size_t)16 * (K_);                                 \
  const __bf16* pb2 = pb1 + (size_t)16 * (K_);                                 \
  const __bf16* pb3 = pb2 + (size_t)16 * (K_);                                 \
  f32x8 acc[2][4] = {};                                                        \
  bf16x16 a0 = ldA(pa0), a1 = ldA(pa1);                                        \
  bf16x16 b0 = ldB(pb0), b1 = ldB(pb1), b2 = ldB(pb2), b3 = ldB(pb3);          \
  for (int k0 = 0; k0 < (K_); k0 += 32) {                                      \
    int kn = (k0 + 32 < (K_)) ? k0 + 32 : 0;                                   \
    bf16x16 na0 = ldA(pa0 + kn), na1 = ldA(pa1 + kn);                          \
    bf16x16 nb0 = ldB(pb0 + kn), nb1 = ldB(pb1 + kn);                          \
    bf16x16 nb2 = ldB(pb2 + kn), nb3 = ldB(pb3 + kn);                          \
    acc[0][0] = wmma_bf16(a0, b0, acc[0][0]);                                  \
    acc[1][0] = wmma_bf16(a1, b0, acc[1][0]);                                  \
    acc[0][1] = wmma_bf16(a0, b1, acc[0][1]);                                  \
    acc[1][1] = wmma_bf16(a1, b1, acc[1][1]);                                  \
    acc[0][2] = wmma_bf16(a0, b2, acc[0][2]);                                  \
    acc[1][2] = wmma_bf16(a1, b2, acc[1][2]);                                  \
    acc[0][3] = wmma_bf16(a0, b3, acc[0][3]);                                  \
    acc[1][3] = wmma_bf16(a1, b3, acc[1][3]);                                  \
    a0 = na0; a1 = na1; b0 = nb0; b1 = nb1; b2 = nb2; b3 = nb3;                \
  }

// ---- GEMM: q = xn @ q_w^T + q_b, scatter to (b,h,tok,d) bf16 ------------
__global__ void k_gemm_q(const __bf16* __restrict__ A, const __bf16* __restrict__ W,
                         const float* __restrict__ bias, __bf16* __restrict__ Q,
                         int M, int K) {
  GEMM_CORE(A, W, M, K)
#pragma unroll
  for (int t = 0; t < 2; ++t)
#pragma unroll
    for (int nt = 0; nt < 4; ++nt) {
      int n = n0 + nt * 16 + l15;
      int hh = n >> 6, d = n & 63;
      float bi = bias[n];
#pragma unroll
      for (int v = 0; v < 8; ++v) {
        int m = m0 + t * 16 + mh + v;
        int bb = m / 784, ml = m % 784;
        Q[(((size_t)(bb * 5 + hh)) * 784 + ml) * 64 + d] = (__bf16)(acc[t][nt][v] + bi);
      }
    }
}

// ---- GEMM: r = relu(bn(xn @ red_w^T)), N=80 (16 per wave), bf16 out -----
__global__ void k_gemm_red(const __bf16* __restrict__ A, const __bf16* __restrict__ W,
                           const float* __restrict__ g, const float* __restrict__ bt,
                           const float* __restrict__ mn, const float* __restrict__ vr,
                           __bf16* __restrict__ R, int M, int K) {
  const int lane = threadIdx.x & 31, wv = threadIdx.x >> 5;
  const int l15 = lane & 15, ka = (lane >> 4) * 8, kb = (lane >> 4) * 8;
  const int mh = (lane >> 4) * 8;
  const int m0 = (blockIdx.x * 8 + wv) * 32;
  const int n0 = blockIdx.y * 16;
  if (m0 >= M) return;
  const __bf16* pa0 = A + (size_t)(m0 + l15) * K + ka;
  const __bf16* pa1 = pa0 + (size_t)16 * K;
  const __bf16* pb0 = W + (size_t)(n0 + l15) * K + kb;
  f32x8 acc[2] = {};
  bf16x16 a0 = ldA(pa0), a1 = ldA(pa1);
  bf16x16 b0 = ldB(pb0);
  for (int k0 = 0; k0 < K; k0 += 32) {
    int kn = (k0 + 32 < K) ? k0 + 32 : 0;
    bf16x16 na0 = ldA(pa0 + kn), na1 = ldA(pa1 + kn);
    bf16x16 nb0 = ldB(pb0 + kn);
    acc[0] = wmma_bf16(a0, b0, acc[0]);
    acc[1] = wmma_bf16(a1, b0, acc[1]);
    a0 = na0; a1 = na1; b0 = nb0;
  }
  int n = n0 + l15;
  float s = g[n] * rsqrtf(vr[n] + 1e-5f);
  float bi = bt[n] - mn[n] * s;
#pragma unroll
  for (int t = 0; t < 2; ++t)
#pragma unroll
    for (int v = 0; v < 8; ++v) {
      int m = m0 + t * 16 + mh + v;
      float val = acc[t][v] * s + bi;
      R[(size_t)m * 80 + n] = (__bf16)(val > 0.f ? val : 0.f);
    }
}

// ---- DWT: r(b,c,28,28) token-major -> coeffs0 (b,14,14, 320) bf16 -------
__global__ void k_dwt(const __bf16* __restrict__ R, __bf16* __restrict__ C0, long total) {
  long i = (long)blockIdx.x * blockDim.x + threadIdx.x;
  if (i >= total) return;
  int cc = i % 80; long t = i / 80;
  int x = t % 14; t /= 14;
  int y = t % 14; int b = t / 14;
  const __bf16* rb = R;
  float a  = (float)rb[((size_t)(b * 784 + (2 * y)     * 28 + 2 * x))     * 80 + cc];
  float bb = (float)rb[((size_t)(b * 784 + (2 * y)     * 28 + 2 * x + 1)) * 80 + cc];
  float c  = (float)rb[((size_t)(b * 784 + (2 * y + 1) * 28 + 2 * x))     * 80 + cc];
  float d  = (float)rb[((size_t)(b * 784 + (2 * y + 1) * 28 + 2 * x + 1)) * 80 + cc];
  size_t base = ((size_t)(b * 196 + y * 14 + x)) * 320 + cc;
  C0[base      ] = (__bf16)((a + bb + c + d) * 0.5f);
  C0[base + 80 ] = (__bf16)((a + bb - c - d) * 0.5f);
  C0[base + 160] = (__bf16)((a - bb + c - d) * 0.5f);
  C0[base + 240] = (__bf16)((a - bb - c + d) * 0.5f);
}

// ---- GEMM: filt 3x3 conv as im2col (K=2880), bn+relu, f32+bf16 out ------
__global__ void k_gemm_filt(const __bf16* __restrict__ C0, const __bf16* __restrict__ W,
                            const float* __restrict__ g, const float* __restrict__ bt,
                            const float* __restrict__ mn, const float* __restrict__ vr,
                            float* __restrict__ CF, __bf16* __restrict__ CB, int M) {
  const int K = 2880;
  const int lane = threadIdx.x & 31, wv = threadIdx.x >> 5;
  const int l15 = lane & 15, ka = (lane >> 4) * 8, kb = (lane >> 4) * 8;
  const int mh = (lane >> 4) * 8;
  const int m0 = (blockIdx.x * 8 + wv) * 32;
  const int n0 = blockIdx.y * 64;
  if (m0 >= M) return;
  int mA = m0 + l15, mB = m0 + 16 + l15;
  int bA = mA / 196, rA = mA % 196, yA = rA / 14, xA = rA % 14;
  int bB = mB / 196, rB = mB % 196, yB = rB / 14, xB = rB % 14;
  const __bf16* pb0 = W + (size_t)(n0 + l15) * K + kb;
  const __bf16* pb1 = pb0 + (size_t)16 * K;
  const __bf16* pb2 = pb1 + (size_t)16 * K;
  const __bf16* pb3 = pb2 + (size_t)16 * K;
  auto loadA = [&](int k, int bb, int yy0, int xx0) -> bf16x16 {
    int s = k / 320, c = k - s * 320;
    int yy = yy0 + s / 3 - 1, xx = xx0 + s % 3 - 1;
    bool valid = (unsigned)yy < 14u && (unsigned)xx < 14u;
    return ldA_pad(C0 + ((size_t)(bb * 196 + yy * 14 + xx) * 320 + c + ka), valid);
  };
  f32x8 acc[2][4] = {};
  bf16x16 a0 = loadA(0, bA, yA, xA), a1 = loadA(0, bB, yB, xB);
  bf16x16 b0 = ldB(pb0), b1 = ldB(pb1), b2 = ldB(pb2), b3 = ldB(pb3);
  for (int k0 = 0; k0 < K; k0 += 32) {
    int kn = (k0 + 32 < K) ? k0 + 32 : 0;
    bf16x16 na0 = loadA(kn, bA, yA, xA), na1 = loadA(kn, bB, yB, xB);
    bf16x16 nb0 = ldB(pb0 + kn), nb1 = ldB(pb1 + kn);
    bf16x16 nb2 = ldB(pb2 + kn), nb3 = ldB(pb3 + kn);
    acc[0][0] = wmma_bf16(a0, b0, acc[0][0]);
    acc[1][0] = wmma_bf16(a1, b0, acc[1][0]);
    acc[0][1] = wmma_bf16(a0, b1, acc[0][1]);
    acc[1][1] = wmma_bf16(a1, b1, acc[1][1]);
    acc[0][2] = wmma_bf16(a0, b2, acc[0][2]);
    acc[1][2] = wmma_bf16(a1, b2, acc[1][2]);
    acc[0][3] = wmma_bf16(a0, b3, acc[0][3]);
    acc[1][3] = wmma_bf16(a1, b3, acc[1][3]);
    a0 = na0; a1 = na1; b0 = nb0; b1 = nb1; b2 = nb2; b3 = nb3;
  }
#pragma unroll
  for (int t = 0; t < 2; ++t)
#pragma unroll
    for (int nt = 0; nt < 4; ++nt) {
      int n = n0 + nt * 16 + l15;
      float s = g[n] * rsqrtf(vr[n] + 1e-5f);
      float bi = bt[n] - mn[n] * s;
#pragma unroll
      for (int v = 0; v < 8; ++v) {
        int m = m0 + t * 16 + mh + v;
        float val = acc[t][nt][v] * s + bi;
        val = val > 0.f ? val : 0.f;
        CF[(size_t)m * 320 + n] = val;
        CB[(size_t)m * 320 + n] = (__bf16)val;
      }
    }
}

// ---- IDWT -> cat columns [320,400) bf16 ---------------------------------
__global__ void k_idwt(const float* __restrict__ CF, __bf16* __restrict__ CAT, long total) {
  long i = (long)blockIdx.x * blockDim.x + threadIdx.x;
  if (i >= total) return;
  int cc = i % 80; long t = i / 80;
  int x = t % 14; t /= 14;
  int y = t % 14; int b = t / 14;
  size_t base = ((size_t)(b * 196 + y * 14 + x)) * 320 + cc;
  float ll = CF[base], lh = CF[base + 80], hl = CF[base + 160], hh = CF[base + 240];
  float a  = (ll + lh + hl + hh) * 0.5f;
  float bv = (ll + lh - hl - hh) * 0.5f;
  float c  = (ll - lh + hl - hh) * 0.5f;
  float d  = (ll - lh - hl + hh) * 0.5f;
  size_t r0 = ((size_t)(b * 784 + (2 * y) * 28 + 2 * x)) * 416 + 320 + cc;
  size_t r1 = ((size_t)(b * 784 + (2 * y + 1) * 28 + 2 * x)) * 416 + 320 + cc;
  CAT[r0]       = (__bf16)a;
  CAT[r0 + 416] = (__bf16)bv;
  CAT[r1]       = (__bf16)c;
  CAT[r1 + 416] = (__bf16)d;
}

// ---- GEMM: kve strided 2x2 conv as im2col (K=1280), +bias, f32 out ------
__global__ void k_gemm_kve(const __bf16* __restrict__ CB, const __bf16* __restrict__ W,
                           const float* __restrict__ bias, float* __restrict__ KV,
                           int M) {
  const int K = 1280;
  const int lane = threadIdx.x & 31, wv = threadIdx.x >> 5;
  const int l15 = lane & 15, ka = (lane >> 4) * 8, kb = (lane >> 4) * 8;
  const int mh = (lane >> 4) * 8;
  const int m0 = (blockIdx.x * 8 + wv) * 32;
  const int n0 = blockIdx.y * 64;
  if (m0 >= M) return;
  int mA = m0 + l15, mB = m0 + 16 + l15;
  int bA = mA / 49, rA = mA % 49, yA = rA / 7, xA = rA % 7;
  int bB = mB / 49, rB = mB % 49, yB = rB / 7, xB = rB % 7;
  const __bf16* pb0 = W + (size_t)(n0 + l15) * K + kb;
  const __bf16* pb1 = pb0 + (size_t)16 * K;
  const __bf16* pb2 = pb1 + (size_t)16 * K;
  const __bf16* pb3 = pb2 + (size_t)16 * K;
  auto loadA = [&](int k, int bb, int yy0, int xx0) -> bf16x16 {
    int s = k / 320, c = k - s * 320;
    int yy = 2 * yy0 + (s >> 1), xx = 2 * xx0 + (s & 1);
    return ldA(CB + ((size_t)(bb * 196 + yy * 14 + xx) * 320 + c + ka));
  };
  f32x8 acc[2][4] = {};
  bf16x16 a0 = loadA(0, bA, yA, xA), a1 = loadA(0, bB, yB, xB);
  bf16x16 b0 = ldB(pb0), b1 = ldB(pb1), b2 = ldB(pb2), b3 = ldB(pb3);
  for (int k0 = 0; k0 < K; k0 += 32) {
    int kn = (k0 + 32 < K) ? k0 + 32 : 0;
    bf16x16 na0 = loadA(kn, bA, yA, xA), na1 = loadA(kn, bB, yB, xB);
    bf16x16 nb0 = ldB(pb0 + kn), nb1 = ldB(pb1 + kn);
    bf16x16 nb2 = ldB(pb2 + kn), nb3 = ldB(pb3 + kn);
    acc[0][0] = wmma_bf16(a0, b0, acc[0][0]);
    acc[1][0] = wmma_bf16(a1, b0, acc[1][0]);
    acc[0][1] = wmma_bf16(a0, b1, acc[0][1]);
    acc[1][1] = wmma_bf16(a1, b1, acc[1][1]);
    acc[0][2] = wmma_bf16(a0, b2, acc[0][2]);
    acc[1][2] = wmma_bf16(a1, b2, acc[1][2]);
    acc[0][3] = wmma_bf16(a0, b3, acc[0][3]);
    acc[1][3] = wmma_bf16(a1, b3, acc[1][3]);
    a0 = na0; a1 = na1; b0 = nb0; b1 = nb1; b2 = nb2; b3 = nb3;
  }
#pragma unroll
  for (int t = 0; t < 2; ++t)
#pragma unroll
    for (int nt = 0; nt < 4; ++nt) {
      int n = n0 + nt * 16 + l15;
      float bi = bias[n];
#pragma unroll
      for (int v = 0; v < 8; ++v) {
        int m = m0 + t * 16 + mh + v;
        KV[(size_t)m * 320 + n] = acc[t][nt][v] + bi;
      }
    }
}

// ---- GEMM: kv = kvsn @ kv_w^T + kv_b, scatter to kpad / vT --------------
__global__ void k_gemm_kv(const __bf16* __restrict__ A, const __bf16* __restrict__ W,
                          const float* __restrict__ bias, __bf16* __restrict__ Kp,
                          __bf16* __restrict__ Vt, int M, int K) {
  GEMM_CORE(A, W, M, K)
#pragma unroll
  for (int t = 0; t < 2; ++t)
#pragma unroll
    for (int nt = 0; nt < 4; ++nt) {
      int n = n0 + nt * 16 + l15;
      int two = n / 320, rem = n % 320;
      int hh = rem >> 6, d = rem & 63;
      float bi = bias[n];
#pragma unroll
      for (int v = 0; v < 8; ++v) {
        int m = m0 + t * 16 + mh + v;
        int bb = m / 49, mm = m % 49;
        float val = acc[t][nt][v] + bi;
        if (two == 0)
          Kp[(((size_t)(bb * 5 + hh)) * 64 + mm) * 64 + d] = (__bf16)val;
        else
          Vt[(((size_t)(bb * 5 + hh)) * 64 + d) * 64 + mm] = (__bf16)val;
      }
    }
}

// ---- Attention: one wave per (b,h, 16-row tile) -------------------------
__global__ void k_attn(const __bf16* __restrict__ Q, const __bf16* __restrict__ Kp,
                       const __bf16* __restrict__ Vt, __bf16* __restrict__ CAT) {
  __shared__ float S[16 * 64];
  __shared__ __bf16 P[16 * 64];
  const int lane = threadIdx.x;
  const int tile = blockIdx.x % 49;
  const int bh = blockIdx.x / 49;
  const int b = bh / 5, h = bh % 5;
  const int m0 = tile * 16;
  const int l15 = lane & 15, ka = (lane >> 4) * 8, kb = (lane >> 4) * 8;
  const int mh = (lane >> 4) * 8;
  const __bf16* q  = Q  + (size_t)bh * 784 * 64;
  const __bf16* kp = Kp + (size_t)bh * 64 * 64;
  const __bf16* vt = Vt + (size_t)bh * 64 * 64;

  // Phase 1: S = q @ k^T  (K=64, fully unrolled, all loads issued up front)
  bf16x16 aq0 = ldA(q + (size_t)(m0 + l15) * 64 + ka);
  bf16x16 aq1 = ldA(q + (size_t)(m0 + l15) * 64 + 32 + ka);
  bf16x16 bk[8];
#pragma unroll
  for (int nt = 0; nt < 4; ++nt) {
    bk[nt * 2]     = ldB(kp + (size_t)(nt * 16 + l15) * 64 + kb);
    bk[nt * 2 + 1] = ldB(kp + (size_t)(nt * 16 + l15) * 64 + 32 + kb);
  }
  f32x8 acc[4] = {};
#pragma unroll
  for (int nt = 0; nt < 4; ++nt) {
    acc[nt] = wmma_bf16(aq0, bk[nt * 2], acc[nt]);
    acc[nt] = wmma_bf16(aq1, bk[nt * 2 + 1], acc[nt]);
  }
#pragma unroll
  for (int nt = 0; nt < 4; ++nt)
#pragma unroll
    for (int v = 0; v < 8; ++v)
      S[(mh + v) * 64 + nt * 16 + l15] = acc[nt][v] * 0.125f;
  __syncthreads();
  // Phase 2: softmax over n<49 per row
  if (lane < 16) {
    float mx = -1e30f;
    for (int n = 0; n < 49; ++n) mx = fmaxf(mx, S[lane * 64 + n]);
    float sum = 0.f;
    for (int n = 0; n < 49; ++n) {
      float e = __expf(S[lane * 64 + n] - mx);
      S[lane * 64 + n] = e;
      sum += e;
    }
    float r = 1.f / sum;
    for (int n = 0; n < 49; ++n) P[lane * 64 + n] = (__bf16)(S[lane * 64 + n] * r);
    for (int n = 49; n < 64; ++n) P[lane * 64 + n] = (__bf16)0.f;
  }
  __syncthreads();
  // Phase 3: out = P @ vT^T
  bf16x16 ap0 = ldA((const __bf16*)P + (size_t)l15 * 64 + ka);
  bf16x16 ap1 = ldA((const __bf16*)P + (size_t)l15 * 64 + 32 + ka);
  bf16x16 bv[8];
#pragma unroll
  for (int nt = 0; nt < 4; ++nt) {
    bv[nt * 2]     = ldB(vt + (size_t)(nt * 16 + l15) * 64 + kb);
    bv[nt * 2 + 1] = ldB(vt + (size_t)(nt * 16 + l15) * 64 + 32 + kb);
  }
  f32x8 o[4] = {};
#pragma unroll
  for (int nt = 0; nt < 4; ++nt) {
    o[nt] = wmma_bf16(ap0, bv[nt * 2], o[nt]);
    o[nt] = wmma_bf16(ap1, bv[nt * 2 + 1], o[nt]);
  }
#pragma unroll
  for (int nt = 0; nt < 4; ++nt)
#pragma unroll
    for (int v = 0; v < 8; ++v) {
      int m = m0 + mh + v;
      int d = nt * 16 + l15;
      CAT[((size_t)(b * 784 + m)) * 416 + h * 64 + d] = (__bf16)(o[nt][v]);
    }
}

// ---- GEMM: xa = x + cat @ proj_w^T + proj_b (f32 out -> d_out) ----------
__global__ void k_gemm_proj(const __bf16* __restrict__ A, const __bf16* __restrict__ W,
                            const float* __restrict__ bias, const float* __restrict__ X,
                            float* __restrict__ O, int M, int K) {
  GEMM_CORE(A, W, M, K)
#pragma unroll
  for (int t = 0; t < 2; ++t)
#pragma unroll
    for (int nt = 0; nt < 4; ++nt) {
      int n = n0 + nt * 16 + l15;
      float bi = bias[n];
#pragma unroll
      for (int v = 0; v < 8; ++v) {
        int m = m0 + t * 16 + mh + v;
        O[(size_t)m * 320 + n] = X[(size_t)m * 320 + n] + acc[t][nt][v] + bi;
      }
    }
}

// ---- GEMM: h1 = xn2 @ fc1_w^T + fc1_b (bf16 out) ------------------------
__global__ void k_gemm_fc1(const __bf16* __restrict__ A, const __bf16* __restrict__ W,
                           const float* __restrict__ bias, __bf16* __restrict__ H,
                           int M, int K) {
  GEMM_CORE(A, W, M, K)
#pragma unroll
  for (int t = 0; t < 2; ++t)
#pragma unroll
    for (int nt = 0; nt < 4; ++nt) {
      int n = n0 + nt * 16 + l15;
      float bi = bias[n];
#pragma unroll
      for (int v = 0; v < 8; ++v) {
        int m = m0 + t * 16 + mh + v;
        H[(size_t)m * 1280 + n] = (__bf16)(acc[t][nt][v] + bi);
      }
    }
}

// ---- depthwise 3x3 + bias + exact GELU (bf16 -> bf16), per chunk --------
__global__ void k_dwconv_gelu(const __bf16* __restrict__ H1, const float* __restrict__ W,
                              const float* __restrict__ B, __bf16* __restrict__ H2,
                              long total) {
  long i = (long)blockIdx.x * blockDim.x + threadIdx.x;
  if (i >= total) return;
  int c = i % 1280; long m = i / 1280;
  int x = m % 28; int y = (m / 28) % 28; int bl = m / 784;
  float s = 0.f;
#pragma unroll
  for (int dy = -1; dy <= 1; ++dy)
#pragma unroll
    for (int dx = -1; dx <= 1; ++dx) {
      int yy = y + dy, xx = x + dx;
      if ((unsigned)yy < 28u && (unsigned)xx < 28u)
        s += (float)H1[((size_t)(bl * 784 + yy * 28 + xx)) * 1280 + c] *
             W[c * 9 + (dy + 1) * 3 + (dx + 1)];
    }
  s += B[c];
  float gel = 0.5f * s * (1.f + erff(s * 0.70710678118654752f));
  H2[(size_t)m * 1280 + c] = (__bf16)gel;
}

// ---- GEMM: out = xa + h2 @ fc2_w^T + fc2_b (in-place on d_out) ----------
__global__ void k_gemm_fc2(const __bf16* __restrict__ A, const __bf16* __restrict__ W,
                           const float* __restrict__ bias, float* __restrict__ O,
                           int M, int K) {
  GEMM_CORE(A, W, M, K)
#pragma unroll
  for (int t = 0; t < 2; ++t)
#pragma unroll
    for (int nt = 0; nt < 4; ++nt) {
      int n = n0 + nt * 16 + l15;
      float bi = bias[n];
#pragma unroll
      for (int v = 0; v < 8; ++v) {
        int m = m0 + t * 16 + mh + v;
        O[(size_t)m * 320 + n] = O[(size_t)m * 320 + n] + acc[t][nt][v] + bi;
      }
    }
}

// ---- host orchestration -------------------------------------------------
extern "C" void kernel_launch(void* const* d_in, const int* in_sizes, int n_in,
                              void* d_out, int out_size, void* d_ws, size_t ws_size,
                              hipStream_t stream) {
  (void)in_sizes; (void)n_in; (void)out_size; (void)ws_size;
  const float* x      = (const float*)d_in[0];
  const float* ln1_g  = (const float*)d_in[1];
  const float* ln1_b  = (const float*)d_in[2];
  const float* q_w    = (const float*)d_in[3];
  const float* q_b    = (const float*)d_in[4];
  const float* red_w  = (const float*)d_in[5];
  const float* red_g  = (const float*)d_in[6];
  const float* red_b  = (const float*)d_in[7];
  const float* red_m  = (const float*)d_in[8];
  const float* red_v  = (const float*)d_in[9];
  const float* filt_w = (const float*)d_in[10];
  const float* filt_g = (const float*)d_in[11];
  const float* filt_b = (const float*)d_in[12];
  const float* filt_m = (const float*)d_in[13];
  const float* filt_v = (const float*)d_in[14];
  const float* kve_w  = (const float*)d_in[15];
  const float* kve_b  = (const float*)d_in[16];
  const float* kvn_g  = (const float*)d_in[17];
  const float* kvn_b  = (const float*)d_in[18];
  const float* kv_w   = (const float*)d_in[19];
  const float* kv_b   = (const float*)d_in[20];
  const float* proj_w = (const float*)d_in[21];
  const float* proj_b = (const float*)d_in[22];
  const float* ln2_g  = (const float*)d_in[23];
  const float* ln2_b  = (const float*)d_in[24];
  const float* fc1_w  = (const float*)d_in[25];
  const float* fc1_b  = (const float*)d_in[26];
  const float* dw_w   = (const float*)d_in[27];
  const float* dw_b   = (const float*)d_in[28];
  const float* fc2_w  = (const float*)d_in[29];
  const float* fc2_b  = (const float*)d_in[30];
  float* out = (float*)d_out;

  char* base = (char*)d_ws;
  size_t off = 0;
  auto take = [&](size_t bytes) -> void* {
    void* p = base + off;
    off += (bytes + 255) & ~(size_t)255;
    return p;
  };
  __bf16* wq    = (__bf16*)take(320L * 320 * 2);
  __bf16* wred  = (__bf16*)take(80L * 320 * 2);
  __bf16* wfilt = (__bf16*)take(320L * 2880 * 2);
  __bf16* wkve  = (__bf16*)take(320L * 1280 * 2);
  __bf16* wkv   = (__bf16*)take(640L * 320 * 2);
  __bf16* wproj = (__bf16*)take(320L * 416 * 2);
  __bf16* wfc1  = (__bf16*)take(1280L * 320 * 2);
  __bf16* wfc2  = (__bf16*)take(320L * 1280 * 2);
  __bf16* xn    = (__bf16*)take(25088L * 320 * 2);  // reused as xn2 after proj
  __bf16* qb    = (__bf16*)take(32L * 5 * 784 * 64 * 2);
  __bf16* rb    = (__bf16*)take(25088L * 80 * 2);
  __bf16* c0    = (__bf16*)take(32L * 196 * 320 * 2);
  float*  cf    = (float*) take(32L * 196 * 320 * 4);
  __bf16* cb    = (__bf16*)take(32L * 196 * 320 * 2);
  float*  kvs   = (float*) take(1568L * 320 * 4);
  __bf16* kvsn  = (__bf16*)take(1568L * 320 * 2);
  __bf16* kp    = (__bf16*)take(32L * 5 * 64 * 64 * 2);
  __bf16* vt    = (__bf16*)take(32L * 5 * 64 * 64 * 2);
  __bf16* cat   = (__bf16*)take(25088L * 416 * 2);
  __bf16* h1    = (__bf16*)take(6272L * 1280 * 2);
  __bf16* h2    = (__bf16*)take(6272L * 1280 * 2);

  auto g1 = [](long n) { return dim3((unsigned)((n + 255) / 256)); };

  // weight packing
  k_f2b<<<g1(320L * 320), 256, 0, stream>>>(q_w, wq, 320L * 320);
  k_f2b<<<g1(80L * 320), 256, 0, stream>>>(red_w, wred, 80L * 320);
  k_pack_filt<<<g1(320L * 320 * 9), 256, 0, stream>>>(filt_w, wfilt);
  k_pack_kve<<<g1(320L * 320 * 4), 256, 0, stream>>>(kve_w, wkve);
  k_f2b<<<g1(640L * 320), 256, 0, stream>>>(kv_w, wkv, 640L * 320);
  k_pack_proj<<<g1(320L * 416), 256, 0, stream>>>(proj_w, wproj);
  k_f2b<<<g1(1280L * 320), 256, 0, stream>>>(fc1_w, wfc1, 1280L * 320);
  k_f2b<<<g1(320L * 1280), 256, 0, stream>>>(fc2_w, wfc2, 320L * 1280);
  // zero pads
  k_zero16<<<g1(32L * 5 * 64 * 64), 256, 0, stream>>>(kp, 32L * 5 * 64 * 64);
  k_zero16<<<g1(32L * 5 * 64 * 64), 256, 0, stream>>>(vt, 32L * 5 * 64 * 64);
  k_zero16<<<g1(25088L * 416), 256, 0, stream>>>(cat, 25088L * 416);

  // LN1
  k_ln<<<dim3(25088 / 8), 256, 0, stream>>>(x, ln1_g, ln1_b, xn, 25088);
  // Q projection
  k_gemm_q<<<dim3(98, 5), 256, 0, stream>>>(xn, wq, q_b, qb, 25088, 320);
  // reduce conv (1x1) + bn + relu
  k_gemm_red<<<dim3(98, 5), 256, 0, stream>>>(xn, wred, red_g, red_b, red_m, red_v,
                                              rb, 25088, 320);
  // DWT
  k_dwt<<<g1(32L * 196 * 80), 256, 0, stream>>>(rb, c0, 32L * 196 * 80);
  // 3x3 filter conv + bn + relu
  k_gemm_filt<<<dim3(25, 5), 256, 0, stream>>>(c0, wfilt, filt_g, filt_b, filt_m,
                                               filt_v, cf, cb, 6272);
  // IDWT -> local part of cat
  k_idwt<<<g1(32L * 196 * 80), 256, 0, stream>>>(cf, cat, 32L * 196 * 80);
  // KV embed strided conv
  k_gemm_kve<<<dim3(7, 5), 256, 0, stream>>>(cb, wkve, kve_b, kvs, 1568);
  // LN on kvs
  k_ln<<<dim3(1568 / 8), 256, 0, stream>>>(kvs, kvn_g, kvn_b, kvsn, 1568);
  // KV projection, scatter to kpad / vT
  k_gemm_kv<<<dim3(7, 10), 256, 0, stream>>>(kvsn, wkv, kv_b, kp, vt, 1568, 320);
  // attention -> cat columns [0,320)
  k_attn<<<dim3(32 * 5 * 49), 32, 0, stream>>>(qb, kp, vt, cat);
  // projection + residual -> d_out (holds xa)
  k_gemm_proj<<<dim3(98, 5), 256, 0, stream>>>(cat, wproj, proj_b, x, out, 25088, 416);
  // LN2 (reuse xn buffer as xn2)
  k_ln<<<dim3(25088 / 8), 256, 0, stream>>>(out, ln2_g, ln2_b, xn, 25088);
  // MLP, batch-chunked (4 chunks of 8 images = 6272 tokens)
  for (int ch = 0; ch < 4; ++ch) {
    const __bf16* a = xn + (size_t)ch * 6272 * 320;
    k_gemm_fc1<<<dim3(25, 20), 256, 0, stream>>>(a, wfc1, fc1_b, h1, 6272, 320);
    k_dwconv_gelu<<<g1(6272L * 1280), 256, 0, stream>>>(h1, dw_w, dw_b, h2,
                                                        6272L * 1280);
    k_gemm_fc2<<<dim3(25, 5), 256, 0, stream>>>(h2, wfc2, fc2_b,
                                                out + (size_t)ch * 6272 * 320,
                                                6272, 1280);
  }
}
